// HMM_10608569221697
// MI455X (gfx1250) — matrix operand order repaired and checked by reading, not compiled, not run on hardware
//
#include <hip/hip_runtime.h>
#include <hip/hip_fp16.h>

typedef __attribute__((ext_vector_type(16))) _Float16 v16h;
typedef __attribute__((ext_vector_type(8)))  float    v8f;
typedef __attribute__((ext_vector_type(8)))  unsigned v8u;

constexpr int BATCH = 256;
constexpr int DIM   = 4096;
constexpr int NS    = 64;

__device__ __forceinline__ unsigned pk2(float a, float b) {
  __half2 h = __floats2half2_rn(a, b);
  unsigned u; __builtin_memcpy(&u, &h, 4); return u;
}
__device__ __forceinline__ float2 unpk2(unsigned u) {
  __half2 h; __builtin_memcpy(&h, &u, 4);
  return __half22float2(h);
}

// One-time setup: row-logsumexp of transitions, linear emission probs, linear
// start probs. ws layout (floats): [0,64)=lseT  [64,128)=E0  [128,192)=E1
// [192,256)=Sv  [256,272)=per-block partial sums.
__global__ void hmm_setup(const float* __restrict__ T,
                          const float* __restrict__ Em,
                          const float* __restrict__ St,
                          float* __restrict__ ws) {
  int i = threadIdx.x;
  if (i >= NS) return;
  float m = -3.4e38f;
  for (int j = 0; j < NS; ++j) m = fmaxf(m, T[i * NS + j]);
  float s = 0.f;
  for (int j = 0; j < NS; ++j) s += expf(T[i * NS + j] - m);
  ws[i] = m + logf(s);                                   // lseT[i]
  float e0 = Em[2 * i], e1 = Em[2 * i + 1];
  float mm = fmaxf(e0, e1);
  float le = mm + logf(expf(e0 - mm) + expf(e1 - mm));
  ws[64 + i]  = expf(e0 - le);                           // E0[i]
  ws[128 + i] = expf(e1 - le);                           // E1[i]
  float ms = -3.4e38f;
  for (int j = 0; j < NS; ++j) ms = fmaxf(ms, St[j]);
  float ss = 0.f;
  for (int j = 0; j < NS; ++j) ss += expf(St[j] - ms);
  ws[192 + i] = expf(St[i] - (ms + logf(ss)));           // Sv[i]
}

// One wave (32 threads) per 16-row batch tile. Scaled linear-space HMM
// forward: per step, q^T = Texp^T @ p^T via 8x v_wmma_f32_16x16x32_f16,
// emission multiply, row-max rescale, shfl_xor(16) repack D -> next B.
__global__ __launch_bounds__(32) void hmm_forward(
    const int* __restrict__ y,
    const float* __restrict__ T,
    const float* __restrict__ ws,
    float* __restrict__ partials) {
  const int  lane = threadIdx.x;
  const int  c    = lane & 15;        // batch column within tile (B/D cols)
  const bool up   = lane >= 16;
  const int  hlf  = up ? 1 : 0;
  const int  b0   = blockIdx.x * 16;

  const float* lseT = ws;
  const float* E0g  = ws + 64;
  const float* E1g  = ws + 128;
  const float* Svg  = ws + 192;

  // ---- A fragments: A = Texp^T (M=state_out, K=state_in), resident f16 ----
  // A layout (16x32 f16): lane holds row M=c; element e -> K = 16*hlf + e
  v16h A[4][2];
#pragma unroll
  for (int m = 0; m < 4; ++m) {
    const int col = m * 16 + c;       // output state = column of Texp
#pragma unroll
    for (int kt = 0; kt < 2; ++kt) {
      v8u au;
#pragma unroll
      for (int d = 0; d < 8; ++d) {
        int e0 = 2 * d;
        int K0 = kt * 32 + 16 * hlf + e0;
        float f0 = expf(T[K0 * NS + col] - lseT[K0]);
        float f1 = expf(T[(K0 + 1) * NS + col] - lseT[K0 + 1]);
        au[d] = pk2(f0, f1);
      }
      A[m][kt] = __builtin_bit_cast(v16h, au);
    }
  }

  // ---- per-lane constants: state j = m*16 + 8*hlf + r (D-row mapping) ----
  float E0c[32], dEc[32], Sc[32];
#pragma unroll
  for (int m = 0; m < 4; ++m)
#pragma unroll
    for (int r = 0; r < 8; ++r) {
      int j = m * 16 + 8 * hlf + r;
      float a = E0g[j], b = E1g[j];
      E0c[m * 8 + r] = a;
      dEc[m * 8 + r] = b - a;
      Sc[m * 8 + r]  = Svg[j];
    }

  const int* yrow = y + (size_t)(b0 + c) * DIM;
  int   ynext = yrow[0];
  float sacc  = 0.f;
  unsigned own[4][4];                 // packed normalized p (f16 pairs)
  v16h B0, B1;

  // Step tail: emission multiply, max-normalize, log-scale, build next B.
  auto tail = [&](float (&vv)[4][8], int t) {
    int yb = ynext;
    int tn = t + 1;
    ynext = yrow[tn < DIM ? tn : DIM - 1];   // prefetch next observation
    float yf = (float)yb;
    float mx = -1e30f;
#pragma unroll
    for (int m = 0; m < 4; ++m)
#pragma unroll
      for (int r = 0; r < 8; ++r) {
        float e = fmaf(yf, dEc[m * 8 + r], E0c[m * 8 + r]);
        float v = vv[m][r] * e;
        vv[m][r] = v;
        mx = fmaxf(mx, v);
      }
    mx = fmaxf(mx, __shfl_xor(mx, 16));      // full per-batch row max
    sacc += __logf(mx);
    float rs = __builtin_amdgcn_rcpf(mx);
#pragma unroll
    for (int m = 0; m < 4; ++m)
#pragma unroll
      for (int i = 0; i < 4; ++i)
        own[m][i] = pk2(vv[m][2 * i] * rs, vv[m][2 * i + 1] * rs);
    unsigned oth[4][4];
#pragma unroll
    for (int m = 0; m < 4; ++m)
#pragma unroll
      for (int i = 0; i < 4; ++i)
        oth[m][i] = (unsigned)__shfl_xor((int)own[m][i], 16);
    // Next B (K=state_in x N=batch): lane holds col c, K = kt*32+16*hlf+e.
    // dwords 0..3 = K rows 0..7 of m-tile (2kt+hlf)  (held by lower lanes)
    // dwords 4..7 = K rows 8..15 of same tile        (held by upper lanes)
#pragma unroll
    for (int kt = 0; kt < 2; ++kt) {
      v8u bu;
#pragma unroll
      for (int i = 0; i < 4; ++i) {
        bu[i]     = up ? oth[2 * kt + 1][i] : own[2 * kt][i];
        bu[4 + i] = up ? own[2 * kt + 1][i] : oth[2 * kt][i];
      }
      if (kt == 0) B0 = __builtin_bit_cast(v16h, bu);
      else         B1 = __builtin_bit_cast(v16h, bu);
    }
  };

  // t = 0: p0 ∝ Sv[j] * E[j, y0]
  {
    float vv[4][8];
#pragma unroll
    for (int m = 0; m < 4; ++m)
#pragma unroll
      for (int r = 0; r < 8; ++r) vv[m][r] = Sc[m * 8 + r];
    tail(vv, 0);
  }

  // t = 1..DIM-1: q^T = Texp^T @ p^T, then shared tail
  for (int t = 1; t < DIM; ++t) {
    float vv[4][8];
#pragma unroll
    for (int m = 0; m < 4; ++m) {
      v8f acc = {0.f, 0.f, 0.f, 0.f, 0.f, 0.f, 0.f, 0.f};
      acc = __builtin_amdgcn_wmma_f32_16x16x32_f16(false, A[m][0], false, B0,
                                                   (short)0, acc, false, false);
      acc = __builtin_amdgcn_wmma_f32_16x16x32_f16(false, A[m][1], false, B1,
                                                   (short)0, acc, false, false);
#pragma unroll
      for (int r = 0; r < 8; ++r) vv[m][r] = acc[r];
    }
    tail(vv, t);
  }

  // log_prob[b] = sacc + log(sum_j p_final[b,j])
  float sum = 0.f;
#pragma unroll
  for (int m = 0; m < 4; ++m)
#pragma unroll
    for (int i = 0; i < 4; ++i) {
      float2 f = unpk2(own[m][i]);
      sum += f.x + f.y;
    }
  sum += __shfl_xor(sum, 16);
  float logp = sacc + __logf(sum);
#pragma unroll
  for (int msk = 1; msk <= 8; msk <<= 1)
    logp += __shfl_xor(logp, msk);      // sum over 16 batch rows of tile
  if (lane == 0) partials[blockIdx.x] = logp;
}

__global__ void hmm_finalize(const float* __restrict__ partials,
                             float* __restrict__ out) {
  if (threadIdx.x == 0) {
    float s = 0.f;
    for (int i = 0; i < BATCH / 16; ++i) s += partials[i];
    out[0] = s * (1.0f / BATCH);
  }
}

extern "C" void kernel_launch(void* const* d_in, const int* in_sizes, int n_in,
                              void* d_out, int out_size, void* d_ws, size_t ws_size,
                              hipStream_t stream) {
  (void)in_sizes; (void)n_in; (void)out_size; (void)ws_size;
  const int*   y  = (const int*)d_in[0];
  const float* T  = (const float*)d_in[1];
  const float* Em = (const float*)d_in[2];
  const float* St = (const float*)d_in[3];
  float* ws = (float*)d_ws;

  hmm_setup<<<1, 64, 0, stream>>>(T, Em, St, ws);
  hmm_forward<<<BATCH / 16, 32, 0, stream>>>(y, T, ws, ws + 256);
  hmm_finalize<<<1, 32, 0, stream>>>(ws + 256, (float*)d_out);
}
